// SelfAttention2D_10393820857050
// MI455X (gfx1250) — matrix operand order repaired and checked by reading, not compiled, not run on hardware
//
#include <hip/hip_runtime.h>
#include <hip/hip_bf16.h>

typedef __attribute__((ext_vector_type(16))) _Float16 v16h;
typedef __attribute__((ext_vector_type(8)))  float    v8f;

#define B_   4
#define CH_  256
#define N_   4096
#define GROUPS_ 8
#define CPG_ 32          // channels per group
#define HEADS_ 4
#define DH_  64          // head dim
#define QKVROWS_ 768

// ---------------------------------------------------------------------------
// WMMA helper (CDNA5 wave32, 16x16x32 f16 -> f32)
// ---------------------------------------------------------------------------
__device__ __forceinline__ v8f wmma_f16(v16h a, v16h b, v8f c) {
  return __builtin_amdgcn_wmma_f32_16x16x32_f16(false, a, false, b, (short)0, c,
                                                false, false);
}

// A fragment (16x32): A[m][k] = src[(m0+m)*ld + k0+k]   (row-major M x K)
// Contiguous: two 8-half runs -> 2x global_load_b128 per lane.
__device__ __forceinline__ v16h load_a(const _Float16* src, int ld, int m0, int k0) {
  const int lane = threadIdx.x & 31;
  const int kb   = (lane < 16) ? 0 : 8;
  const _Float16* p = src + (size_t)(m0 + (lane & 15)) * ld + k0 + kb;
  v16h a;
#pragma unroll
  for (int i = 0; i < 8; ++i)  a[i] = p[i];        // K = kb + i
#pragma unroll
  for (int i = 8; i < 16; ++i) a[i] = p[i + 8];    // K = kb + 16 + (i-8)
  return a;
}

// B fragment (32x16) from transposed source: B[k][n] = src[(n0+n)*ld + k0+k]
// Two 8-half runs in k (K = kb + i, kb + 16 + (i-8)) -> 2x global_load_b128 per lane.
__device__ __forceinline__ v16h load_b_t(const _Float16* src, int ld, int k0, int n0) {
  const int lane = threadIdx.x & 31;
  const int kb   = (lane < 16) ? 0 : 8;
  const _Float16* p = src + (size_t)(n0 + (lane & 15)) * ld + k0 + kb;
  v16h b;
#pragma unroll
  for (int hi = 0; hi < 8; ++hi) b[hi] = p[hi];
#pragma unroll
  for (int hi = 8; hi < 16; ++hi) b[hi] = p[hi + 8];
  return b;
}

// ---------------------------------------------------------------------------
// Kernel 1: convert weights fp32 -> f16
// ---------------------------------------------------------------------------
__global__ void __launch_bounds__(256)
cvt_w_kernel(const float* __restrict__ qkv_w, const float* __restrict__ proj_w,
             _Float16* __restrict__ wq16, _Float16* __restrict__ wp16) {
  int i = blockIdx.x * blockDim.x + threadIdx.x;
  if (i < QKVROWS_ * CH_) wq16[i] = (_Float16)qkv_w[i];
  if (i < CH_ * CH_)      wp16[i] = (_Float16)proj_w[i];
}

// ---------------------------------------------------------------------------
// Kernel 2: GroupNorm statistics (one block per (b, group))
// ---------------------------------------------------------------------------
__global__ void __launch_bounds__(256)
gn_stats_kernel(const float* __restrict__ x, float* __restrict__ stats) {
  const int bg = blockIdx.x;                      // b*GROUPS + g
  const float* px = x + (size_t)bg * CPG_ * N_;   // group channels contiguous
  float s = 0.f, s2 = 0.f;
  for (int i = threadIdx.x; i < CPG_ * N_; i += 256) {
    float v = px[i];
    s += v; s2 += v * v;
  }
  __shared__ float sh0[256], sh1[256];
  sh0[threadIdx.x] = s; sh1[threadIdx.x] = s2;
  __syncthreads();
  for (int o = 128; o > 0; o >>= 1) {
    if (threadIdx.x < o) {
      sh0[threadIdx.x] += sh0[threadIdx.x + o];
      sh1[threadIdx.x] += sh1[threadIdx.x + o];
    }
    __syncthreads();
  }
  if (threadIdx.x == 0) {
    const float inv_n = 1.0f / (float)(CPG_ * N_);
    float mean = sh0[0] * inv_n;
    float var  = sh1[0] * inv_n - mean * mean;
    stats[bg * 2 + 0] = mean;
    stats[bg * 2 + 1] = rsqrtf(var + 1e-5f);
  }
}

// ---------------------------------------------------------------------------
// Kernel 3: GroupNorm apply -> f16 h^T stored [b][s][c]  (grid: B * N/16)
// ---------------------------------------------------------------------------
__global__ void __launch_bounds__(256)
gn_apply_kernel(const float* __restrict__ x, const float* __restrict__ gamma,
                const float* __restrict__ beta, const float* __restrict__ stats,
                _Float16* __restrict__ ht) {
  const int b  = blockIdx.x / (N_ / 16);
  const int s0 = (blockIdx.x % (N_ / 16)) * 16;
  const int c  = threadIdx.x;
  const int g  = c / CPG_;
  const float mean = stats[(b * GROUPS_ + g) * 2 + 0];
  const float rstd = stats[(b * GROUPS_ + g) * 2 + 1];
  const float sc = gamma[c] * rstd;
  const float sh = beta[c] - mean * sc;
  const float* px = x + ((size_t)b * CH_ + c) * N_ + s0;
  _Float16*   pt = ht + ((size_t)b * N_ + s0) * CH_ + c;
#pragma unroll
  for (int i = 0; i < 16; ++i)
    pt[(size_t)i * CH_] = (_Float16)(px[i] * sc + sh);
}

// ---------------------------------------------------------------------------
// Kernel 4: qkv GEMM  (per batch: W[768x256] x h[256x4096])
// Each wave computes a 32x64 strip: 2 A-frags x 4 B-frags -> 8 WMMA per K-step.
//   Q part -> qt [b][s][256], K part -> kt [b][s][256], V part -> v16 [b][c][N]
// grid (64, 3, B)
// ---------------------------------------------------------------------------
__global__ void __launch_bounds__(256)
qkv_gemm_kernel(const _Float16* __restrict__ wq16, const float* __restrict__ qkv_b,
                const _Float16* __restrict__ ht, _Float16* __restrict__ qt,
                _Float16* __restrict__ kt, _Float16* __restrict__ v16) {
  const int b    = blockIdx.z;
  const int wave = threadIdx.x >> 5;
  const int m0   = (blockIdx.y * 8 + wave) * 32;  // output rows (o), 32 per wave
  const int n0   = blockIdx.x * 64;               // output cols (s)
  const _Float16* hb = ht + (size_t)b * N_ * CH_;

  v8f acc[2][4] = {};
  for (int k0 = 0; k0 < CH_; k0 += 32) {
    const v16h a0 = load_a(wq16, CH_, m0, k0);
    const v16h a1 = load_a(wq16, CH_, m0 + 16, k0);
#pragma unroll
    for (int t = 0; t < 4; ++t) {
      const v16h bf = load_b_t(hb, CH_, k0, n0 + 16 * t);
      acc[0][t] = wmma_f16(a0, bf, acc[0][t]);
      acc[1][t] = wmma_f16(a1, bf, acc[1][t]);
    }
  }
  const int lane = threadIdx.x & 31;
  const int n  = lane & 15;
  const int mb = (lane < 16) ? 0 : 8;
  const int part = m0 >> 8;          // 32-row tile stays inside one 256-row part
  const int c0   = (m0 & 255) + mb;

  if (part < 2) {
    _Float16* dst = (part == 0 ? qt : kt) + (size_t)b * N_ * CH_;
#pragma unroll
    for (int u = 0; u < 2; ++u)
#pragma unroll
      for (int t = 0; t < 4; ++t) {
        const int s = n0 + 16 * t + n;
        _Float16* p = dst + (size_t)s * CH_ + c0 + 16 * u;
#pragma unroll
        for (int r = 0; r < 8; ++r)  // 8 consecutive channels
          p[r] = (_Float16)(acc[u][t][r] + qkv_b[m0 + 16 * u + mb + r]);
      }
  } else {
    _Float16* dst = v16 + (size_t)b * CH_ * N_;
#pragma unroll
    for (int u = 0; u < 2; ++u)
#pragma unroll
      for (int t = 0; t < 4; ++t) {
        const int s = n0 + 16 * t + n;
#pragma unroll
        for (int r = 0; r < 8; ++r)
          dst[(size_t)(c0 + 16 * u + r) * N_ + s] =
              (_Float16)(acc[u][t][r] + qkv_b[m0 + 16 * u + mb + r]);
      }
  }
}

// ---------------------------------------------------------------------------
// Kernel 5: flash attention per (b, head). grid (32, 4, B); wave -> 16 q-rows.
// Software-pipelined: next K-frags and this chunk's V-frags are issued before
// the softmax VALU chain, which hides their latency.
// ---------------------------------------------------------------------------
__global__ void __launch_bounds__(256)
attn_kernel(const _Float16* __restrict__ qt, const _Float16* __restrict__ kt,
            const _Float16* __restrict__ v16, _Float16* __restrict__ at) {
  const int b    = blockIdx.z;
  const int hd   = blockIdx.y;
  const int wave = threadIdx.x >> 5;
  const int lane = threadIdx.x & 31;
  const int i0   = (blockIdx.x * 8 + wave) * 16;

  const _Float16* Qh = qt + (size_t)b * N_ * CH_ + hd * DH_;    // [s][c] ld=256
  const _Float16* Kh = kt + (size_t)b * N_ * CH_ + hd * DH_;    // [s][c] ld=256
  const _Float16* Vh = v16 + ((size_t)b * CH_ + hd * DH_) * N_; // [d][j] ld=N

  __shared__ _Float16 lds[8][16 * 32];   // per-wave P tile (16 x 32 f16)
  _Float16* pw = lds[wave];

  // Loop-invariant Q fragments: A[m=i][k=d] = Qh[(i0+i)*256 + d]
  const v16h aq0 = load_a(Qh, CH_, i0, 0);
  const v16h aq1 = load_a(Qh, CH_, i0, 32);

  const int n  = lane & 15;
  const int mb = (lane < 16) ? 0 : 8;

  float m[8], l[8];
#pragma unroll
  for (int r = 0; r < 8; ++r) { m[r] = -1e30f; l[r] = 0.f; }
  v8f o0 = {}, o1 = {}, o2 = {}, o3 = {};

  // preload first K-chunk fragments
  v16h kb0 = load_b_t(Kh, CH_, 0,  0);
  v16h kb1 = load_b_t(Kh, CH_, 32, 0);
  v16h kb2 = load_b_t(Kh, CH_, 0,  16);
  v16h kb3 = load_b_t(Kh, CH_, 32, 16);

  for (int j0 = 0; j0 < N_; j0 += 32) {
    // S = (Q^T K) * 1/sqrt(d) : two 16x16 f32 tiles (cols j0.., j0+16..)
    v8f s0 = {}, s1 = {};
    s0 = wmma_f16(aq0, kb0, s0);
    s0 = wmma_f16(aq1, kb1, s0);
    s1 = wmma_f16(aq0, kb2, s1);
    s1 = wmma_f16(aq1, kb3, s1);

    // Issue this chunk's V fragments now (consumed after softmax + relayout)
    const v16h vb0 = load_b_t(Vh, N_, j0,  0);
    const v16h vb1 = load_b_t(Vh, N_, j0, 16);
    const v16h vb2 = load_b_t(Vh, N_, j0, 32);
    const v16h vb3 = load_b_t(Vh, N_, j0, 48);

    // Issue next chunk's K fragments (consumed next iteration)
    if (j0 + 32 < N_) {
      kb0 = load_b_t(Kh, CH_, 0,  j0 + 32);
      kb1 = load_b_t(Kh, CH_, 32, j0 + 32);
      kb2 = load_b_t(Kh, CH_, 0,  j0 + 48);
      kb3 = load_b_t(Kh, CH_, 32, j0 + 48);
      __builtin_prefetch(Vh + j0 + 64, 0, 1);   // global_prefetch_b8
    }

    // online softmax: row M lives across the 16 lanes of a half-wave
    float corr[8];
#pragma unroll
    for (int r = 0; r < 8; ++r) {
      float v0 = s0[r] * 0.125f;     // 1/sqrt(64)
      float v1 = s1[r] * 0.125f;
      float rm = fmaxf(v0, v1);
#pragma unroll
      for (int off = 8; off > 0; off >>= 1)
        rm = fmaxf(rm, __shfl_xor(rm, off, 16));
      float nm = fmaxf(m[r], rm);
      corr[r] = __expf(m[r] - nm);
      float p0 = __expf(v0 - nm);
      float p1 = __expf(v1 - nm);
      s0[r] = p0; s1[r] = p1;
      float rs = p0 + p1;
#pragma unroll
      for (int off = 8; off > 0; off >>= 1)
        rs += __shfl_xor(rs, off, 16);
      l[r] = l[r] * corr[r] + rs;
      m[r] = nm;
    }
#pragma unroll
    for (int r = 0; r < 8; ++r) {
      o0[r] *= corr[r]; o1[r] *= corr[r]; o2[r] *= corr[r]; o3[r] *= corr[r];
    }

    // D-layout -> A-layout relayout of P via per-wave LDS (same-wave DS in order)
#pragma unroll
    for (int r = 0; r < 8; ++r) {
      pw[(mb + r) * 32 + n]      = (_Float16)s0[r];
      pw[(mb + r) * 32 + 16 + n] = (_Float16)s1[r];
    }
    v16h ap;
    {
      const int kb = (lane < 16) ? 0 : 8;
      const _Float16* pp = pw + (lane & 15) * 32 + kb;
#pragma unroll
      for (int i = 0; i < 8; ++i)  ap[i] = pp[i];
#pragma unroll
      for (int i = 8; i < 16; ++i) ap[i] = pp[i + 8];
    }

    // O^T[i, d] += P[i, j] * V^T[j, d]
    o0 = wmma_f16(ap, vb0, o0);
    o1 = wmma_f16(ap, vb1, o1);
    o2 = wmma_f16(ap, vb2, o2);
    o3 = wmma_f16(ap, vb3, o3);
  }

  // epilogue: divide by l, store out^T as [s][c] -> at[b][i][hd*64 + d]
  _Float16* outp = at + (size_t)b * N_ * CH_ + hd * DH_;
  v8f oacc[4] = {o0, o1, o2, o3};
#pragma unroll
  for (int t = 0; t < 4; ++t) {
#pragma unroll
    for (int r = 0; r < 8; ++r) {
      float v = oacc[t][r] / l[r];
      outp[(size_t)(i0 + mb + r) * CH_ + t * 16 + n] = (_Float16)v;
    }
  }
}

// ---------------------------------------------------------------------------
// Kernel 6: proj GEMM + bias + residual -> fp32 out. grid (64, 1, B)
// Each wave computes a 32x64 strip (8 WMMA per K-step).
// ---------------------------------------------------------------------------
__global__ void __launch_bounds__(256)
proj_gemm_kernel(const _Float16* __restrict__ wp16, const float* __restrict__ proj_b,
                 const _Float16* __restrict__ at, const float* __restrict__ x,
                 float* __restrict__ out) {
  const int b    = blockIdx.z;
  const int wave = threadIdx.x >> 5;
  const int m0   = wave * 32;                     // 8 waves x 32 = 256 rows
  const int n0   = blockIdx.x * 64;
  const _Float16* ab = at + (size_t)b * N_ * CH_;

  v8f acc[2][4] = {};
  for (int k0 = 0; k0 < CH_; k0 += 32) {
    const v16h a0 = load_a(wp16, CH_, m0, k0);
    const v16h a1 = load_a(wp16, CH_, m0 + 16, k0);
#pragma unroll
    for (int t = 0; t < 4; ++t) {
      const v16h bf = load_b_t(ab, CH_, k0, n0 + 16 * t);
      acc[0][t] = wmma_f16(a0, bf, acc[0][t]);
      acc[1][t] = wmma_f16(a1, bf, acc[1][t]);
    }
  }
  const int lane = threadIdx.x & 31;
  const int n  = lane & 15;
  const int mb = (lane < 16) ? 0 : 8;
  const float* xb = x   + (size_t)b * CH_ * N_;
  float*       ob = out + (size_t)b * CH_ * N_;
#pragma unroll
  for (int u = 0; u < 2; ++u)
#pragma unroll
    for (int t = 0; t < 4; ++t) {
#pragma unroll
      for (int r = 0; r < 8; ++r) {
        int o = m0 + 16 * u + mb + r;
        size_t idx = (size_t)o * N_ + n0 + 16 * t + n;
        ob[idx] = acc[u][t][r] + proj_b[o] + xb[idx];
      }
    }
}

// ---------------------------------------------------------------------------
extern "C" void kernel_launch(void* const* d_in, const int* in_sizes, int n_in,
                              void* d_out, int out_size, void* d_ws, size_t ws_size,
                              hipStream_t stream) {
  const float* x      = (const float*)d_in[0];
  const float* norm_w = (const float*)d_in[1];
  const float* norm_b = (const float*)d_in[2];
  const float* qkv_w  = (const float*)d_in[3];
  const float* qkv_b  = (const float*)d_in[4];
  const float* proj_w = (const float*)d_in[5];
  const float* proj_b = (const float*)d_in[6];
  float* out = (float*)d_out;

  char* ws = (char*)d_ws;
  _Float16* ht   = (_Float16*)ws;  ws += (size_t)B_ * N_ * CH_ * sizeof(_Float16);
  _Float16* qt   = (_Float16*)ws;  ws += (size_t)B_ * N_ * CH_ * sizeof(_Float16);
  _Float16* kt   = (_Float16*)ws;  ws += (size_t)B_ * N_ * CH_ * sizeof(_Float16);
  _Float16* v16  = (_Float16*)ws;  ws += (size_t)B_ * CH_ * N_ * sizeof(_Float16);
  _Float16* at   = (_Float16*)ws;  ws += (size_t)B_ * N_ * CH_ * sizeof(_Float16);
  _Float16* wq16 = (_Float16*)ws;  ws += (size_t)QKVROWS_ * CH_ * sizeof(_Float16);
  _Float16* wp16 = (_Float16*)ws;  ws += (size_t)CH_ * CH_ * sizeof(_Float16);
  float*    stats = (float*)ws;    // B*GROUPS*2 floats

  cvt_w_kernel<<<(QKVROWS_ * CH_ + 255) / 256, 256, 0, stream>>>(qkv_w, proj_w, wq16, wp16);
  gn_stats_kernel<<<B_ * GROUPS_, 256, 0, stream>>>(x, stats);
  gn_apply_kernel<<<B_ * (N_ / 16), 256, 0, stream>>>(x, norm_w, norm_b, stats, ht);
  qkv_gemm_kernel<<<dim3(N_ / 64, QKVROWS_ / 256, B_), 256, 0, stream>>>(wq16, qkv_b, ht, qt, kt, v16);
  attn_kernel<<<dim3(N_ / 16 / 8, HEADS_, B_), 256, 0, stream>>>(qt, kt, v16, at);
  proj_gemm_kernel<<<dim3(N_ / 64, 1, B_), 256, 0, stream>>>(wp16, proj_b, at, x, out);
}
